// EMAVectorQuantizer_41549513621534
// MI455X (gfx1250) — matrix operand order repaired and checked by reading, not compiled, not run on hardware
//
#include <hip/hip_runtime.h>
#include <hip/hip_bf16.h>
#include <stdint.h>

// ---------------------------------------------------------------------------
// VQ (vector-quantizer) forward on gfx1250 using v_wmma_f32_16x16x32_bf16.
// dot[n,k] via split-bf16 (hi+lo) GEMM -> fp32-accurate argmin.
// ---------------------------------------------------------------------------

typedef __attribute__((ext_vector_type(16))) __bf16 v16bf;
typedef __attribute__((ext_vector_type(8)))  float  v8f;

#define N_POINTS    65536
#define NUM_CODES   4096
#define EMBED_DIM   64
#define K_TILES     (NUM_CODES / 16)          // 256 code tiles
#define WAVES_PB    8
#define BLOCKS_MAIN (N_POINTS / 16 / WAVES_PB) // 512

__device__ __forceinline__ __bf16 f2bf(float f) {
  uint32_t u = __builtin_bit_cast(uint32_t, f);
  uint32_t r = u + 0x7FFFu + ((u >> 16) & 1u);   // round-to-nearest-even
  return __builtin_bit_cast(__bf16, (uint16_t)(r >> 16));
}
__device__ __forceinline__ float bf2f(__bf16 h) {
  uint32_t u = ((uint32_t)__builtin_bit_cast(uint16_t, h)) << 16;
  return __builtin_bit_cast(float, u);
}

// ---------------------------------------------------------------------------
// Pack codebook into WMMA B-operand layout, split into bf16 hi/lo.
// Layout: block index g = ((t*4 + s)*32 + lane), s: 0=hi/c0 1=hi/c1 2=lo/c0 3=lo/c1
// B element i of lane: n = lane&15, K(within 32-chunk) = 16*(lane>>4) + i.
// ---------------------------------------------------------------------------
__global__ void vq_pack_b(const float* __restrict__ emb, __bf16* __restrict__ Bpk) {
  int g    = blockIdx.x * blockDim.x + threadIdx.x;   // 32768 threads
  int lane = g & 31;
  int s    = (g >> 5) & 3;
  int t    = g >> 7;
  int chunk = s & 1;
  int isLo  = s >> 1;
  int n     = lane & 15;
  int halfK = lane >> 4;
  const float* src = emb + ((size_t)(t * 16 + n)) * EMBED_DIM + chunk * 32 + halfK * 16;
  v16bf o;
#pragma unroll
  for (int i = 0; i < 16; ++i) {
    float f  = src[i];
    __bf16 h = f2bf(f);
    o[i] = isLo ? f2bf(f - bf2f(h)) : h;
  }
  *(v16bf*)(Bpk + (size_t)g * 16) = o;
}

// |e_k|^2 and zero the histogram (must be re-zeroed every launch).
__global__ void vq_prep(const float* __restrict__ emb, float* __restrict__ e_sq,
                        unsigned int* __restrict__ counts) {
  int k = blockIdx.x * blockDim.x + threadIdx.x;      // 4096 threads
  const float* row = emb + (size_t)k * EMBED_DIM;
  float ssum = 0.f;
#pragma unroll 8
  for (int d = 0; d < EMBED_DIM; ++d) ssum += row[d] * row[d];
  e_sq[k]   = ssum;
  counts[k] = 0u;
}

// ---------------------------------------------------------------------------
// Main kernel: 512 blocks x 256 threads (8 waves). Wave handles 16 rows.
// ---------------------------------------------------------------------------
__global__ void __launch_bounds__(256) vq_main(
    const float* __restrict__ z_e, const float* __restrict__ emb,
    const __bf16* __restrict__ Bpk, const float* __restrict__ e_sq,
    float* __restrict__ out_zq, float* __restrict__ out_idx,
    unsigned int* __restrict__ counts, float* __restrict__ block_loss) {

  __shared__ __align__(16) float zt[WAVES_PB][16 * EMBED_DIM];  // 32 KB
  __shared__ float red_v[WAVES_PB][16][16];                     // 8 KB
  __shared__ int   red_i[WAVES_PB][16][16];                     // 8 KB
  __shared__ int   rowidx[WAVES_PB][16];
  __shared__ float lpart[WAVES_PB][32];

  const int tid  = threadIdx.x;
  const int w    = tid >> 5;
  const int lane = tid & 31;
  const int half = lane >> 4;
  const int nl   = lane & 15;
  const int row0 = (blockIdx.x * WAVES_PB + w) * 16;

  // ---- stage 16x64 fp32 z tile into LDS (coalesced b128) ----
  {
    const float4* src = (const float4*)(z_e + (size_t)row0 * EMBED_DIM);
    float4* dst = (float4*)&zt[w][0];
#pragma unroll
    for (int j = 0; j < 8; ++j) dst[j * 32 + lane] = src[j * 32 + lane];
  }
  __syncthreads();

  // ---- build A operands (contiguous k, same order as B pack): m = lane&15; K = 16*half + i ----
  v16bf ah0, ah1, al0, al1;
  {
    const float* zrow = &zt[w][nl * EMBED_DIM];
    float fa[2][16];
#pragma unroll
    for (int c = 0; c < 2; ++c) {
      float4 q0 = *(const float4*)(zrow + c * 32 + half * 16);
      float4 q1 = *(const float4*)(zrow + c * 32 + half * 16 + 4);
      float4 q2 = *(const float4*)(zrow + c * 32 + half * 16 + 8);
      float4 q3 = *(const float4*)(zrow + c * 32 + half * 16 + 12);
      fa[c][0]=q0.x;  fa[c][1]=q0.y;  fa[c][2]=q0.z;  fa[c][3]=q0.w;
      fa[c][4]=q1.x;  fa[c][5]=q1.y;  fa[c][6]=q1.z;  fa[c][7]=q1.w;
      fa[c][8]=q2.x;  fa[c][9]=q2.y;  fa[c][10]=q2.z; fa[c][11]=q2.w;
      fa[c][12]=q3.x; fa[c][13]=q3.y; fa[c][14]=q3.z; fa[c][15]=q3.w;
    }
#pragma unroll
    for (int i = 0; i < 16; ++i) {
      __bf16 h0 = f2bf(fa[0][i]);
      __bf16 h1 = f2bf(fa[1][i]);
      ah0[i] = h0; al0[i] = f2bf(fa[0][i] - bf2f(h0));
      ah1[i] = h1; al1[i] = f2bf(fa[1][i] - bf2f(h1));
    }
  }

  // ---- argmin over 256 code tiles; 6x wmma_bf16 (split-precision) ----
  float minv[8];
  int   mini[8];
#pragma unroll
  for (int r = 0; r < 8; ++r) { minv[r] = 3.4e38f; mini[r] = 0x7fffffff; }

  const v16bf* BV = (const v16bf*)Bpk;
  for (int t = 0; t < K_TILES; ++t) {
    size_t base = (size_t)t * 128 + lane;
    v16bf bh0 = BV[base];
    v16bf bh1 = BV[base + 32];
    v16bf bl0 = BV[base + 64];
    v16bf bl1 = BV[base + 96];
    v8f acc = {};
    acc = __builtin_amdgcn_wmma_f32_16x16x32_bf16(false, ah0, false, bh0, (short)0, acc, false, false);
    acc = __builtin_amdgcn_wmma_f32_16x16x32_bf16(false, ah1, false, bh1, (short)0, acc, false, false);
    acc = __builtin_amdgcn_wmma_f32_16x16x32_bf16(false, ah0, false, bl0, (short)0, acc, false, false);
    acc = __builtin_amdgcn_wmma_f32_16x16x32_bf16(false, ah1, false, bl1, (short)0, acc, false, false);
    acc = __builtin_amdgcn_wmma_f32_16x16x32_bf16(false, al0, false, bh0, (short)0, acc, false, false);
    acc = __builtin_amdgcn_wmma_f32_16x16x32_bf16(false, al1, false, bh1, (short)0, acc, false, false);
    int n = t * 16 + nl;
    float es = e_sq[n];
#pragma unroll
    for (int r = 0; r < 8; ++r) {
      float d = es - 2.0f * acc[r];   // |z|^2 is row-constant: irrelevant for argmin
      if (d < minv[r]) { minv[r] = d; mini[r] = n; }
    }
  }

  // ---- cross-lane argmin reduction (D layout: m = r + 8*half, n = lane&15) ----
#pragma unroll
  for (int r = 0; r < 8; ++r) {
    int m = r + half * 8;
    red_v[w][m][nl] = minv[r];
    red_i[w][m][nl] = mini[r];
  }
  __syncthreads();

  if (lane < 16) {
    float best = 3.4e38f;
    int bidx = 0x7fffffff;
#pragma unroll
    for (int c = 0; c < 16; ++c) {
      float v = red_v[w][lane][c];
      int   i = red_i[w][lane][c];
      if (v < best || (v == best && i < bidx)) { best = v; bidx = i; }
    }
    rowidx[w][lane] = bidx;
    out_idx[row0 + lane] = (float)bidx;
    atomicAdd(&counts[bidx], 1u);
  }
  __syncthreads();

  // ---- gather z_q, write straight-through output, accumulate loss partial ----
  float lacc = 0.f;
#pragma unroll
  for (int m = 0; m < 16; ++m) {
    int idx = rowidx[w][m];
    float2 e2 = *(const float2*)(emb + (size_t)idx * EMBED_DIM + lane * 2);
    float2 z2 = *(const float2*)(&zt[w][m * EMBED_DIM + lane * 2]);
    float dx = e2.x - z2.x;
    float dy = e2.y - z2.y;
    float2 st;                   // z_e + (z_q - z_e): mimic reference arithmetic
    st.x = z2.x + dx;
    st.y = z2.y + dy;
    *(float2*)(out_zq + (size_t)(row0 + m) * EMBED_DIM + lane * 2) = st;
    lacc += dx * dx + dy * dy;
  }
  lpart[w][lane] = lacc;
  __syncthreads();

  if (tid == 0) {                // fixed-order deterministic block partial
    float s = 0.f;
    for (int ww = 0; ww < WAVES_PB; ++ww)
      for (int l = 0; l < 32; ++l) s += lpart[ww][l];
    block_loss[blockIdx.x] = s;
  }
}

// ---------------------------------------------------------------------------
// Scalars: commitment loss, perplexity, n_active (deterministic, single thread)
// ---------------------------------------------------------------------------
__global__ void vq_final(const unsigned int* __restrict__ counts,
                         const float* __restrict__ block_loss,
                         float* __restrict__ out_scal) {
  if (blockIdx.x == 0 && threadIdx.x == 0) {
    float tot = 0.f;
    for (int b = 0; b < BLOCKS_MAIN; ++b) tot += block_loss[b];
    float loss = 0.25f * tot / ((float)N_POINTS * (float)EMBED_DIM);
    float H = 0.f;
    int act = 0;
    for (int k = 0; k < NUM_CODES; ++k) {
      float p = (float)counts[k] * (1.0f / (float)N_POINTS);
      H += p * logf(p + 1e-10f);
      act += (p > 0.001f) ? 1 : 0;
    }
    out_scal[0] = loss;
    out_scal[1] = expf(-H);
    out_scal[2] = (float)act;
  }
}

// ---------------------------------------------------------------------------
extern "C" void kernel_launch(void* const* d_in, const int* in_sizes, int n_in,
                              void* d_out, int out_size, void* d_ws, size_t ws_size,
                              hipStream_t stream) {
  const float* z_e = (const float*)d_in[0];       // [65536, 64] fp32
  const float* emb = (const float*)d_in[1];       // [4096, 64]  fp32

  float* out     = (float*)d_out;
  float* out_zq  = out;                           // 65536*64
  float* out_idx = out + (size_t)N_POINTS * EMBED_DIM;   // 65536
  float* out_sc  = out_idx + N_POINTS;            // 3 scalars

  char* ws = (char*)d_ws;
  __bf16*       Bpk        = (__bf16*)ws;                         // 1,048,576 B
  float*        e_sq       = (float*)(ws + 1048576);              // 16 KB
  unsigned int* counts     = (unsigned int*)(ws + 1048576 + 16384);
  float*        block_loss = (float*)(ws + 1048576 + 32768);      // 2 KB

  vq_pack_b<<<128, 256, 0, stream>>>(emb, Bpk);
  vq_prep  <<<16, 256, 0, stream>>>(emb, e_sq, counts);
  vq_main  <<<BLOCKS_MAIN, 256, 0, stream>>>(z_e, emb, Bpk, e_sq,
                                             out_zq, out_idx, counts, block_loss);
  vq_final <<<1, 32, 0, stream>>>(counts, block_loss, out_sc);
}
